// Block_89498528514674
// MI455X (gfx1250) — compile-verified, matrix operand order checked by reading, not run on hardware
//
#include <hip/hip_runtime.h>

// ---------------------------------------------------------------------------
// Problem constants (from reference): B=2, T=2048, C=768, H=12, D=64, E=9
// ---------------------------------------------------------------------------
#define CDIM  768
#define FF4   3072
#define FF2   1536
#define NEXP  9
#define HEADS 12
#define DHEAD 64
#define BB    2
#define TT    2048
#define MTOK  (BB * TT)          // 4096 tokens

typedef __bf16 bf16;
typedef __attribute__((ext_vector_type(16))) __bf16 v16bf;
typedef __attribute__((ext_vector_type(8)))  float  v8f;
typedef __attribute__((ext_vector_type(4)))  unsigned int u32x4;
typedef __attribute__((ext_vector_type(8)))  int i32x8;
typedef __attribute__((ext_vector_type(4)))  int i32x4;

#if defined(__has_builtin)
#if __has_builtin(__builtin_amdgcn_tensor_load_to_lds) && __has_builtin(__builtin_amdgcn_s_wait_tensorcnt)
#define HAVE_TDM 1
#endif
#endif

// ---------------------------------------------------------------------------
// Device helpers
// ---------------------------------------------------------------------------
__device__ __forceinline__ v8f wmma_bf16(v16bf a, v16bf b, v8f c) {
  // v_wmma_f32_16x16x32_bf16  (8 args: neg_a, A, neg_b, B, c_mod, C, reuse_a, reuse_b)
  return __builtin_amdgcn_wmma_f32_16x16x32_bf16(false, a, false, b, (short)0, c, false, false);
}

// 16-bit operand K-index map for wave32 WMMA fragments (ISA 7.12.2):
// lane-group g = lane/16; element e (0..15) -> K = (e/8)*16 + g*8 + (e%8)
// Elements form two contiguous 8-element runs: [g*8, g*8+8) and [16+g*8, ...).
__device__ __forceinline__ int kmap(int e, int g) {
  return ((e >> 3) << 4) + (g << 3) + (e & 7);
}

// Load a 16-element fragment from a row of 32 contiguous bf16 (stride-1 K):
// becomes two 16-byte (b128) LDS/global loads.
__device__ __forceinline__ v16bf load_frag_row(const bf16* __restrict__ row, int g) {
  v16bf f;
#pragma unroll
  for (int e = 0; e < 16; ++e) f[e] = row[kmap(e, g)];
  return f;
}

__device__ __forceinline__ float apply_act(float x, int act) {
  if (act == 1) return 0.5f * x * (1.0f + erff(x * 0.70710678118654752f)); // exact gelu
  if (act == 2) return x / (1.0f + __expf(-x));                           // silu
  return x;
}

__device__ __forceinline__ float block_reduce_sum256(float v, float* red) {
  int tid = threadIdx.x;
  red[tid] = v;
  __syncthreads();
  for (int s = 128; s > 0; s >>= 1) {
    if (tid < s) red[tid] += red[tid + s];
    __syncthreads();
  }
  float r = red[0];
  __syncthreads();
  return r;
}

__device__ __forceinline__ unsigned pack_bf16x2(bf16 lo, bf16 hi) {
  unsigned l = __builtin_bit_cast(unsigned short, lo);
  unsigned h = __builtin_bit_cast(unsigned short, hi);
  return l | (h << 16);
}

// ---------------------------------------------------------------------------
// fp32 -> bf16 convert (weights + activations)
// ---------------------------------------------------------------------------
__global__ __launch_bounds__(256) void k_convert(const float* __restrict__ in,
                                                 bf16* __restrict__ out, long long n) {
  long long i = (long long)blockIdx.x * 256 + threadIdx.x;
  if (i < n) out[i] = (bf16)in[i];
}

// ---------------------------------------------------------------------------
// LN1 + structural embedding, writes bf16 activations for the QKV GEMM
// ---------------------------------------------------------------------------
__global__ __launch_bounds__(256) void k_ln_embed(const float* __restrict__ x,
                                                  const int* __restrict__ idx,
                                                  const float* __restrict__ g,
                                                  const float* __restrict__ b,
                                                  const float* __restrict__ emb,
                                                  bf16* __restrict__ outB) {
  __shared__ float red[256];
  __shared__ float buf[CDIM];
  long long t = blockIdx.x;
  const float* xr = x + t * CDIM;
  float s = 0.f;
  for (int c = threadIdx.x; c < CDIM; c += 256) { float v = xr[c]; buf[c] = v; s += v; }
  float mean = block_reduce_sum256(s, red) * (1.0f / CDIM);
  float vs = 0.f;
  for (int c = threadIdx.x; c < CDIM; c += 256) { float d = buf[c] - mean; vs += d * d; }
  float var = block_reduce_sum256(vs, red) * (1.0f / CDIM);
  float rstd = rsqrtf(var + 1e-5f);
  int id = idx[t];
  int sid = (id >= 1 && id <= 3) ? id : 0;
  const float* er = emb + sid * CDIM;
  for (int c = threadIdx.x; c < CDIM; c += 256) {
    float v = (buf[c] - mean) * rstd * g[c] + b[c] + er[c];
    outB[t * CDIM + c] = (bf16)v;
  }
}

// ---------------------------------------------------------------------------
// Generic LayerNorm (+optional residual add input, +optional activation),
// writes f32 and/or bf16.  F <= FF4, blockDim = 256, one row per block.
// ---------------------------------------------------------------------------
__global__ __launch_bounds__(256) void k_ln_act(const float* __restrict__ in,
                                                const float* __restrict__ add,
                                                const float* __restrict__ g,
                                                const float* __restrict__ b,
                                                int F, int act,
                                                float* __restrict__ outF,
                                                bf16* __restrict__ outB) {
  __shared__ float red[256];
  __shared__ float buf[FF4];
  long long t = blockIdx.x;
  const float* r = in + t * F;
  const float* a = add ? add + t * F : nullptr;
  float s = 0.f;
  for (int c = threadIdx.x; c < F; c += 256) {
    float v = r[c] + (a ? a[c] : 0.f);
    buf[c] = v; s += v;
  }
  float mean = block_reduce_sum256(s, red) / (float)F;
  float vs = 0.f;
  for (int c = threadIdx.x; c < F; c += 256) { float d = buf[c] - mean; vs += d * d; }
  float var = block_reduce_sum256(vs, red) / (float)F;
  float rstd = rsqrtf(var + 1e-5f);
  for (int c = threadIdx.x; c < F; c += 256) {
    float v = (buf[c] - mean) * rstd * g[c] + b[c];
    v = apply_act(v, act);
    if (outF) outF[t * F + c] = v;
    if (outB) outB[t * F + c] = (bf16)v;
  }
}

// ---------------------------------------------------------------------------
// Tiled bf16 WMMA GEMM:  out = act(A[M,K] @ B[K,N] + bias)
// Block tile 128x128, 256 threads = 8 waves (4 M-waves x 2 N-waves),
// wave tile 32x64 -> 8 WMMAs per 32-deep K step.
// A tile: row-major LDS [128][32+8]; optionally DMA'd by the Tensor Data
//   Mover (with LDS pad descriptor: 4 dwords every 16 dwords).
// B tile: stored TRANSPOSED in LDS as packed bf16x2 k-pairs so B fragments
//   load as two ds_load_b128 per fragment (no scalar u16 gathers).
// ---------------------------------------------------------------------------
#define BM 128
#define BN 128
#define KT 32

__global__ __launch_bounds__(256)
void k_gemm_bf16(const bf16* __restrict__ A, int lda,
                 const bf16* __restrict__ Bw, int ldb,
                 const float* __restrict__ bias,
                 float* __restrict__ outF, bf16* __restrict__ outB, int ldc,
                 int M, int N, int K, int act, int use_tdm) {
  __shared__ bf16     As[BM][KT + 8];           // 128 x 40 bf16
  __shared__ unsigned BsT32[BN][KT / 2 + 4];    // col-major B, packed k-pairs

  const int tid   = threadIdx.x;
  const int lane  = tid & 31;
  const int wid   = tid >> 5;
  const int waveM = wid & 3;   // 4 waves over M (32 rows each)
  const int waveN = wid >> 2;  // 2 waves over N (64 cols each)
  const int m0    = blockIdx.y * BM;
  const int n0    = blockIdx.x * BN;
  const int g     = lane >> 4;
  const int ln    = lane & 15;

  v8f acc[2][4];
#pragma unroll
  for (int mi = 0; mi < 2; ++mi)
#pragma unroll
    for (int ni = 0; ni < 4; ++ni)
#pragma unroll
      for (int v = 0; v < 8; ++v) acc[mi][ni][v] = 0.f;

  const int arow = tid >> 1, acol = (tid & 1) * 16;  // 128x32 A tile: 16 bf16/thread
  const int kp = tid & 15;      // k-pair index (k = 2*kp, 2*kp+1)
  const int cg = tid >> 4;      // column group (8 columns each)

  for (int kt = 0; kt < K; kt += KT) {
    if (kt + KT < K)  // global_prefetch of the next A tile
      __builtin_prefetch(A + (size_t)(m0 + arow) * lda + kt + KT + acol, 0, 1);

#ifdef HAVE_TDM
    if (use_tdm) {
      if (wid == 0) {
        // D#: 2D tile [BM x KT] of bf16 from row-major A, padded into LDS:
        // row = 32 bf16 = 16 dwords (pad_interval code 3), pad = 4 dwords (code 3).
        unsigned long long ga = (unsigned long long)(const void*)(A + (size_t)m0 * lda + kt);
        unsigned ldsa = (unsigned)(unsigned long long)(&As[0][0]);
        u32x4 g0 = { 1u,                                       // count=1
                     ldsa,                                     // lds_addr
                     (unsigned)ga,                             // global_addr[31:0]
                     (unsigned)((ga >> 32) & 0x01FFFFFFull) | 0x80000000u }; // addr | type=2
        i32x8 g1;
        g1[0] = (int)((1u << 16) |        // data_size = 2 bytes
                      (1u << 20) |        // pad_enable
                      (3u << 22) |        // pad_interval: every 16 dwords
                      (3u << 25));        // pad_amount: 4 dwords
        g1[1] = (int)(((unsigned)K & 0xFFFFu) << 16);                           // tensor_dim0 lo
        g1[2] = (int)((((unsigned)K >> 16) & 0xFFFFu) | (((unsigned)M & 0xFFFFu) << 16)); // dim0 hi | dim1 lo
        g1[3] = (int)((((unsigned)M >> 16) & 0xFFFFu) | ((unsigned)KT << 16));  // dim1 hi | tile_dim0
        g1[4] = BM;        // tile_dim1
        g1[5] = lda;       // tensor_dim0_stride[31:0]
        g1[6] = 0;
        g1[7] = 0;
        i32x4 z4 = {0, 0, 0, 0};
        i32x8 z8;
#pragma unroll
        for (int q = 0; q < 8; ++q) z8[q] = 0;
        __builtin_amdgcn_tensor_load_to_lds(g0, g1, z4, z4, z8, 0);
        __builtin_amdgcn_s_wait_tensorcnt(0);
      }
    } else
#endif
    { // cooperative A tile load (16 contiguous bf16 = two b128 per thread)
      const bf16* src = A + (size_t)(m0 + arow) * lda + kt + acol;
#pragma unroll
      for (int e = 0; e < 16; ++e) As[arow][acol + e] = src[e];
    }

    { // cooperative B tile load, transposed into packed k-pairs:
      // thread reads rows k=2kp,2kp+1 at 8 contiguous columns, writes 8 b32.
      const bf16* r0 = Bw + (size_t)(kt + 2 * kp) * ldb + n0 + cg * 8;
      const bf16* r1 = r0 + ldb;
#pragma unroll
      for (int j = 0; j < 8; ++j)
        BsT32[cg * 8 + j][kp] = pack_bf16x2(r0[j], r1[j]);
    }
    __syncthreads();

    v16bf afr[2], bfr[4];
#pragma unroll
    for (int mi = 0; mi < 2; ++mi)
      afr[mi] = load_frag_row(&As[waveM * 32 + mi * 16 + ln][0], g);
#pragma unroll
    for (int ni = 0; ni < 4; ++ni)
      bfr[ni] = load_frag_row((const bf16*)&BsT32[waveN * 64 + ni * 16 + ln][0], g);
#pragma unroll
    for (int mi = 0; mi < 2; ++mi)
#pragma unroll
      for (int ni = 0; ni < 4; ++ni)
        acc[mi][ni] = wmma_bf16(afr[mi], bfr[ni], acc[mi][ni]);
    __syncthreads();
  }

  // Epilogue: C layout = VGPR v holds row (v + 8*lanegroup), col = lane%16
#pragma unroll
  for (int mi = 0; mi < 2; ++mi) {
#pragma unroll
    for (int ni = 0; ni < 4; ++ni) {
      int rbase = m0 + waveM * 32 + mi * 16;
      int col   = n0 + waveN * 64 + ni * 16 + ln;
      float bv  = bias ? bias[col] : 0.f;
#pragma unroll
      for (int v = 0; v < 8; ++v) {
        int row = rbase + v + 8 * g;
        float xv = apply_act(acc[mi][ni][v] + bv, act);
        size_t o = (size_t)row * ldc + col;
        if (outF) outF[o] = xv;
        if (outB) outB[o] = (bf16)xv;
      }
    }
  }
}

// ---------------------------------------------------------------------------
// Split fused qkv [M, 3C] into per-head Q/K/V [B,H,T,64] bf16
// ---------------------------------------------------------------------------
__global__ __launch_bounds__(256) void k_split_heads(const bf16* __restrict__ qkv,
                                                     bf16* __restrict__ Q,
                                                     bf16* __restrict__ Kk,
                                                     bf16* __restrict__ V) {
  long long i = (long long)blockIdx.x * 256 + threadIdx.x;
  if (i >= (long long)MTOK * CDIM) return;
  int tk = (int)(i / CDIM), c = (int)(i % CDIM);
  int h = c >> 6, d = c & 63;
  int b = tk / TT, t = tk % TT;
  size_t dst = (((size_t)(b * HEADS + h)) * TT + t) * DHEAD + d;
  size_t srow = (size_t)tk * (3 * CDIM);
  Q[dst]  = qkv[srow + c];
  Kk[dst] = qkv[srow + CDIM + c];
  V[dst]  = qkv[srow + 2 * CDIM + c];
}

// ---------------------------------------------------------------------------
// Flash-style attention, one wave (32 threads) per (head, 16-query tile).
// QK^T and P@V both via v_wmma_f32_16x16x32_bf16.  Online softmax with
// qk clamp to [-30,30] and scale 1/sqrt(64).  V tiles staged transposed in
// LDS (packed k-pairs) so vb fragments load as two ds_load_b128 each.
// ---------------------------------------------------------------------------
__global__ __launch_bounds__(32) void k_attn(const bf16* __restrict__ Q,
                                             const bf16* __restrict__ Kk,
                                             const bf16* __restrict__ V,
                                             bf16* __restrict__ attB) {
  __shared__ float    Sm[16][32];
  __shared__ bf16     Pm[16][32];
  __shared__ unsigned VsT32[DHEAD][KT / 2 + 4];  // transposed V tile, packed k-pairs
  __shared__ float    alphas[16];
  __shared__ float    rstat[16];

  const int lane = threadIdx.x;
  const int g    = lane >> 4;
  const int ln   = lane & 15;
  const int bh   = blockIdx.y;
  const int b    = bh / HEADS;
  const int h    = bh % HEADS;
  const int q0   = blockIdx.x * 16;
  const size_t base = (size_t)bh * TT * DHEAD;
  const bf16* Qg = Q + base;
  const bf16* Kg = Kk + base;
  const bf16* Vg = V + base;

  // Q fragments (A operand, 16x64 split into two 16x32)
  v16bf qa0, qa1;
  {
    const bf16* p = Qg + (size_t)(q0 + ln) * DHEAD;
    qa0 = load_frag_row(p, g);
    qa1 = load_frag_row(p + 32, g);
  }

  v8f o[4];
#pragma unroll
  for (int f = 0; f < 4; ++f)
#pragma unroll
    for (int v = 0; v < 8; ++v) o[f][v] = 0.f;
  float mrow = -1e30f, lrow = 0.f;

  for (int j0 = 0; j0 < TT; j0 += 32) {
    { // stage V tile transposed: lane (kp, half) copies 2 key rows x 32 d
      const int kp = ln;        // key pair 0..15
      const int half = g;       // d half: 0..1 (32 d each)
      const bf16* r0 = Vg + (size_t)(j0 + 2 * kp) * DHEAD + half * 32;
      const bf16* r1 = r0 + DHEAD;
#pragma unroll
      for (int d = 0; d < 32; ++d)
        VsT32[half * 32 + d][kp] = pack_bf16x2(r0[d], r1[d]);
    }

    // S = clamp(Q K^T / 8) for 32 keys -> two 16x16 frags
#pragma unroll
    for (int nf = 0; nf < 2; ++nf) {
      v8f c;
#pragma unroll
      for (int v = 0; v < 8; ++v) c[v] = 0.f;
      const bf16* p = Kg + (size_t)(j0 + nf * 16 + ln) * DHEAD;
      v16bf kb0 = load_frag_row(p, g);
      v16bf kb1 = load_frag_row(p + 32, g);
      c = wmma_bf16(qa0, kb0, c);
      c = wmma_bf16(qa1, kb1, c);
#pragma unroll
      for (int v = 0; v < 8; ++v) {
        float s = c[v] * 0.125f;
        s = fminf(fmaxf(s, -30.f), 30.f);
        Sm[v + 8 * g][nf * 16 + ln] = s;
      }
    }
    __syncthreads();

    if (lane < 16) { // per-row online softmax update (threads 0..15 own rows)
      int r = lane;
      float tmax = -1e30f;
      for (int j = 0; j < 32; ++j) tmax = fmaxf(tmax, Sm[r][j]);
      float mnew = fmaxf(mrow, tmax);
      float al = __expf(mrow - mnew);
      float sum = 0.f;
      for (int j = 0; j < 32; ++j) {
        float p = __expf(Sm[r][j] - mnew);
        Pm[r][j] = (bf16)p;
        sum += p;
      }
      lrow = lrow * al + sum;
      mrow = mnew;
      alphas[r] = al;
    }
    __syncthreads();

    // rescale accumulators and add P @ V (A 16x32, B 32x64 -> 4 frags)
    v16bf pa = load_frag_row(&Pm[ln][0], g);
#pragma unroll
    for (int v = 0; v < 8; ++v) {
      float al = alphas[v + 8 * g];
#pragma unroll
      for (int f = 0; f < 4; ++f) o[f][v] *= al;
    }
#pragma unroll
    for (int nf = 0; nf < 4; ++nf) {
      v16bf vb = load_frag_row((const bf16*)&VsT32[nf * 16 + ln][0], g);
      o[nf] = wmma_bf16(pa, vb, o[nf]);
    }
    __syncthreads();
  }

  if (lane < 16) rstat[lane] = 1.0f / (lrow + 1e-10f);
  __syncthreads();

#pragma unroll
  for (int f = 0; f < 4; ++f)
#pragma unroll
    for (int v = 0; v < 8; ++v) {
      int row = v + 8 * g;
      int col = f * 16 + ln;
      float val = o[f][v] * rstat[row];
      attB[((size_t)(b * TT + q0 + row)) * CDIM + h * DHEAD + col] = (bf16)val;
    }
}

// ---------------------------------------------------------------------------
// Router: scores = qr . type_q[e], noise gate, top-2, masked softmax gates
// ---------------------------------------------------------------------------
__global__ __launch_bounds__(256) void k_router(const float* __restrict__ qr,
                                                const float* __restrict__ h2,
                                                const float* __restrict__ type_q,
                                                const float* __restrict__ nw,
                                                const float* __restrict__ nb,
                                                const float* __restrict__ temp,
                                                const float* __restrict__ noise,
                                                float* __restrict__ gates) {
  __shared__ float red[256];
  __shared__ float sc[12];
  long long t = blockIdx.x;
  const float* q  = qr + t * CDIM;
  const float* hh = h2 + t * CDIM;
  for (int e = 0; e < NEXP + 1; ++e) {
    const float* w = (e < NEXP) ? (type_q + (size_t)e * CDIM) : nw;
    const float* v = (e < NEXP) ? q : hh;
    float s = 0.f;
    for (int c = threadIdx.x; c < CDIM; c += 256) s += v[c] * w[c];
    float r = block_reduce_sum256(s, red);
    if (threadIdx.x == 0) sc[e] = r;
    __syncthreads();
  }
  if (threadIdx.x == 0) {
    float te = temp[0] * 0.9025f;           // 0.95^(T//1000), T=2048
    te = fminf(fmaxf(te, 0.1f), 1.0f);
    float sig = 1.0f / (1.0f + __expf(-(sc[NEXP] + nb[0])));
    float ny[NEXP];
    for (int e = 0; e < NEXP; ++e) ny[e] = sc[e] + te * noise[t * NEXP + e] * sig;
    int i1 = 0;
    for (int e = 1; e < NEXP; ++e) if (ny[e] > ny[i1]) i1 = e;
    int i2 = -1;
    for (int e = 0; e < NEXP; ++e) if (e != i1 && (i2 < 0 || ny[e] > ny[i2])) i2 = e;
    float inv_t = 1.0f / (te + 1e-6f);
    float mx = fmaxf(ny[i1] * inv_t, ny[i2] * inv_t);
    float p[NEXP], den = 0.f;
    for (int e = 0; e < NEXP; ++e) {
      float lg = (e == i1 || e == i2) ? ny[e] * inv_t : -1e30f;
      p[e] = __expf(lg - mx);
      den += p[e];
    }
    for (int e = 0; e < NEXP; ++e) gates[t * NEXP + e] = p[e] / den;
  }
}

// ---------------------------------------------------------------------------
// Elementwise helpers
// ---------------------------------------------------------------------------
__global__ __launch_bounds__(256) void k_add(const float* __restrict__ a,
                                             const float* __restrict__ b,
                                             float* __restrict__ out, long long n) {
  long long i = (long long)blockIdx.x * 256 + threadIdx.x;
  if (i < n) out[i] = a[i] + b[i];
}

__global__ __launch_bounds__(256) void k_scale_accum(float* __restrict__ acc,
                                                     const float* __restrict__ y,
                                                     const float* __restrict__ gates,
                                                     int ge, long long n) {
  long long i = (long long)blockIdx.x * 256 + threadIdx.x;
  if (i < n) {
    long long t = i / CDIM;
    acc[i] += gates[t * NEXP + ge] * y[i];
  }
}

// ---------------------------------------------------------------------------
// Host orchestration
// ---------------------------------------------------------------------------
extern "C" void kernel_launch(void* const* d_in, const int* in_sizes, int n_in,
                              void* d_out, int out_size, void* d_ws, size_t ws_size,
                              hipStream_t stream) {
  (void)in_sizes; (void)n_in; (void)out_size; (void)ws_size;
  // Inputs in setup_inputs() order; params flattened in dict insertion order.
  const float* x       = (const float*)d_in[0];
  const int*   idx     = (const int*)d_in[1];
  const float* noise   = (const float*)d_in[2];
  const float* ln1_g   = (const float*)d_in[3];
  const float* ln1_b   = (const float*)d_in[4];
  const float* ln2_g   = (const float*)d_in[5];
  const float* ln2_b   = (const float*)d_in[6];
  const float* semb    = (const float*)d_in[7];
  const float* qkv_w   = (const float*)d_in[8];
  const float* out_w   = (const float*)d_in[9];
  const float* out_b   = (const float*)d_in[10];
  const float* rn_w1   = (const float*)d_in[11];
  const float* rn_b1   = (const float*)d_in[12];
  const float* rn_w2   = (const float*)d_in[13];
  const float* rn_b2   = (const float*)d_in[14];
  const float* type_q  = (const float*)d_in[15];
  const float* noise_w = (const float*)d_in[16];
  const float* noise_b = (const float*)d_in[17];
  const float* temp    = (const float*)d_in[18];
  const float* d_w1 = (const float*)d_in[19]; const float* d_b1 = (const float*)d_in[20];
  const float* d_w2 = (const float*)d_in[21]; const float* d_b2 = (const float*)d_in[22];
  const float* d_lg = (const float*)d_in[23]; const float* d_lb = (const float*)d_in[24];
  const float* d_w3 = (const float*)d_in[25]; const float* d_b3 = (const float*)d_in[26];
  const float* d_ng = (const float*)d_in[27]; const float* d_nb = (const float*)d_in[28];
  const float* w_w1 = (const float*)d_in[29]; const float* w_b1 = (const float*)d_in[30];
  const float* w_lg = (const float*)d_in[31]; const float* w_lb = (const float*)d_in[32];
  const float* w_w2 = (const float*)d_in[33]; const float* w_b2 = (const float*)d_in[34];
  const float* w_ng = (const float*)d_in[35]; const float* w_nb = (const float*)d_in[36];
  const float* h_a1 = (const float*)d_in[37]; const float* h_ab1 = (const float*)d_in[38];
  const float* h_a2 = (const float*)d_in[39]; const float* h_ab2 = (const float*)d_in[40];
  const float* h_b1 = (const float*)d_in[41]; const float* h_bb1 = (const float*)d_in[42];
  const float* h_b2 = (const float*)d_in[43]; const float* h_bb2 = (const float*)d_in[44];
  const float* h_pw = (const float*)d_in[45]; const float* h_pb = (const float*)d_in[46];
  const float* h_ng = (const float*)d_in[47]; const float* h_nb = (const float*)d_in[48];

  char* wsb = (char*)d_ws;
  size_t off = 0;
  auto alloc = [&](size_t bytes) -> void* {
    void* p = wsb + off;
    off = (off + bytes + 255) & ~(size_t)255;
    return p;
  };

  const long long M = MTOK;
  // bf16 weight copies
  bf16* qkv_wb = (bf16*)alloc((size_t)CDIM * 3 * CDIM * 2);
  bf16* out_wb = (bf16*)alloc((size_t)CDIM * CDIM * 2);
  bf16* rn_w1b = (bf16*)alloc((size_t)CDIM * FF4 * 2);
  bf16* rn_w2b = (bf16*)alloc((size_t)FF4 * CDIM * 2);
  bf16* dw1b = (bf16*)alloc((size_t)3 * CDIM * FF4 * 2);
  bf16* dw2b = (bf16*)alloc((size_t)3 * FF4 * FF4 * 2);
  bf16* dw3b = (bf16*)alloc((size_t)3 * FF4 * CDIM * 2);
  bf16* ww1b = (bf16*)alloc((size_t)3 * CDIM * FF4 * 2);
  bf16* ww2b = (bf16*)alloc((size_t)3 * FF4 * CDIM * 2);
  bf16* ha1b = (bf16*)alloc((size_t)3 * CDIM * FF4 * 2);
  bf16* ha2b = (bf16*)alloc((size_t)3 * FF4 * CDIM * 2);
  bf16* hb1b = (bf16*)alloc((size_t)3 * CDIM * FF2 * 2);
  bf16* hb2b = (bf16*)alloc((size_t)3 * FF2 * CDIM * 2);
  bf16* hpwb = (bf16*)alloc((size_t)3 * FF2 * CDIM * 2);
  // activations
  bf16*  hA_b   = (bf16*)alloc((size_t)M * CDIM * 2);
  bf16*  qkvb   = (bf16*)alloc((size_t)M * 3 * CDIM * 2);
  bf16*  Qb     = (bf16*)alloc((size_t)M * CDIM * 2);
  bf16*  Kb     = (bf16*)alloc((size_t)M * CDIM * 2);
  bf16*  Vb     = (bf16*)alloc((size_t)M * CDIM * 2);
  bf16*  attB   = (bf16*)alloc((size_t)M * CDIM * 2);
  float* proj_f = (float*)alloc((size_t)M * CDIM * 4);
  float* x1_f   = (float*)alloc((size_t)M * CDIM * 4);
  float* h2_f   = (float*)alloc((size_t)M * CDIM * 4);
  bf16*  h2_b   = (bf16*)alloc((size_t)M * CDIM * 2);
  float* qr_f   = (float*)alloc((size_t)M * CDIM * 4);
  float* gates  = (float*)alloc((size_t)M * NEXP * 4);
  float* tF4_f  = (float*)alloc((size_t)M * FF4 * 4);
  bf16*  tF4_b  = (bf16*)alloc((size_t)M * FF4 * 2);
  float* tC_f   = (float*)alloc((size_t)M * CDIM * 4);
  float* y_f    = (float*)alloc((size_t)M * CDIM * 4);
  bf16*  pcat_b = (bf16*)alloc((size_t)M * 2 * CDIM * 2);
  float* moe    = (float*)alloc((size_t)M * CDIM * 4);

  auto conv = [&](const float* s, bf16* d, long long n) {
    k_convert<<<(unsigned)((n + 255) / 256), 256, 0, stream>>>(s, d, n);
  };
  auto gemm = [&](const bf16* A, int lda, const bf16* Bw, int ldb, const float* bias,
                  float* oF, bf16* oB, int ldc, int Md, int Nd, int Kd, int act, int tdm) {
    dim3 grid(Nd / BN, Md / BM);
    k_gemm_bf16<<<grid, 256, 0, stream>>>(A, lda, Bw, ldb, bias, oF, oB, ldc,
                                          Md, Nd, Kd, act, tdm);
  };
  auto lnact = [&](const float* in, const float* add, const float* g, const float* b,
                   int F, int act, float* oF, bf16* oB) {
    k_ln_act<<<(unsigned)M, 256, 0, stream>>>(in, add, g, b, F, act, oF, oB);
  };

  // -- weight conversion (deterministic; idempotent each call) --
  conv(qkv_w, qkv_wb, (long long)CDIM * 3 * CDIM);
  conv(out_w, out_wb, (long long)CDIM * CDIM);
  conv(rn_w1, rn_w1b, (long long)CDIM * FF4);
  conv(rn_w2, rn_w2b, (long long)FF4 * CDIM);
  conv(d_w1, dw1b, 3LL * CDIM * FF4);
  conv(d_w2, dw2b, 3LL * FF4 * FF4);
  conv(d_w3, dw3b, 3LL * FF4 * CDIM);
  conv(w_w1, ww1b, 3LL * CDIM * FF4);
  conv(w_w2, ww2b, 3LL * FF4 * CDIM);
  conv(h_a1, ha1b, 3LL * CDIM * FF4);
  conv(h_a2, ha2b, 3LL * FF4 * CDIM);
  conv(h_b1, hb1b, 3LL * CDIM * FF2);
  conv(h_b2, hb2b, 3LL * FF2 * CDIM);
  conv(h_pw, hpwb, 3LL * FF2 * CDIM);

  const long long nMC = M * CDIM;
  const unsigned gMC = (unsigned)((nMC + 255) / 256);

  // -- attention branch --
  k_ln_embed<<<(unsigned)M, 256, 0, stream>>>(x, idx, ln1_g, ln1_b, semb, hA_b);
  gemm(hA_b, CDIM, qkv_wb, 3 * CDIM, nullptr, nullptr, qkvb, 3 * CDIM,
       MTOK, 3 * CDIM, CDIM, 0, 0);
  k_split_heads<<<gMC, 256, 0, stream>>>(qkvb, Qb, Kb, Vb);
  k_attn<<<dim3(TT / 16, BB * HEADS), 32, 0, stream>>>(Qb, Kb, Vb, attB);
  gemm(attB, CDIM, out_wb, CDIM, out_b, proj_f, nullptr, CDIM, MTOK, CDIM, CDIM, 0, 0);
  k_add<<<gMC, 256, 0, stream>>>(x, proj_f, x1_f, nMC);

  // -- MoE branch --
  lnact(x1_f, nullptr, ln2_g, ln2_b, CDIM, 0, h2_f, h2_b);
  gemm(h2_b, CDIM, rn_w1b, FF4, rn_b1, nullptr, tF4_b, FF4, MTOK, FF4, CDIM, 1, 1);
  gemm(tF4_b, FF4, rn_w2b, CDIM, rn_b2, qr_f, nullptr, CDIM, MTOK, CDIM, FF4, 0, 1);
  k_router<<<(unsigned)M, 256, 0, stream>>>(qr_f, h2_f, type_q, noise_w, noise_b,
                                            temp, noise, gates);
  (void)hipMemsetAsync(moe, 0, (size_t)nMC * 4, stream);

  for (int e = 0; e < 3; ++e) {  // deep experts
    gemm(h2_b, CDIM, dw1b + (size_t)e * CDIM * FF4, FF4, d_b1 + e * FF4,
         nullptr, tF4_b, FF4, MTOK, FF4, CDIM, 2, 1);
    gemm(tF4_b, FF4, dw2b + (size_t)e * FF4 * FF4, FF4, d_b2 + e * FF4,
         tF4_f, nullptr, FF4, MTOK, FF4, FF4, 0, 1);
    lnact(tF4_f, nullptr, d_lg + e * FF4, d_lb + e * FF4, FF4, 2, nullptr, tF4_b);
    gemm(tF4_b, FF4, dw3b + (size_t)e * FF4 * CDIM, CDIM, d_b3 + e * CDIM,
         tC_f, nullptr, CDIM, MTOK, CDIM, FF4, 0, 1);
    lnact(tC_f, h2_f, d_ng + e * CDIM, d_nb + e * CDIM, CDIM, 0, y_f, nullptr);
    k_scale_accum<<<gMC, 256, 0, stream>>>(moe, y_f, gates, e, nMC);
  }
  for (int e = 0; e < 3; ++e) {  // wide experts
    gemm(h2_b, CDIM, ww1b + (size_t)e * CDIM * FF4, FF4, w_b1 + e * FF4,
         tF4_f, nullptr, FF4, MTOK, FF4, CDIM, 1, 1);
    lnact(tF4_f, nullptr, w_lg + e * FF4, w_lb + e * FF4, FF4, 0, nullptr, tF4_b);
    gemm(tF4_b, FF4, ww2b + (size_t)e * FF4 * CDIM, CDIM, w_b2 + e * CDIM,
         tC_f, nullptr, CDIM, MTOK, CDIM, FF4, 0, 1);
    lnact(tC_f, h2_f, w_ng + e * CDIM, w_nb + e * CDIM, CDIM, 0, y_f, nullptr);
    k_scale_accum<<<gMC, 256, 0, stream>>>(moe, y_f, gates, 3 + e, nMC);
  }
  for (int e = 0; e < 3; ++e) {  // hybrid experts
    gemm(h2_b, CDIM, ha1b + (size_t)e * CDIM * FF4, FF4, h_ab1 + e * FF4,
         nullptr, tF4_b, FF4, MTOK, FF4, CDIM, 1, 1);
    gemm(tF4_b, FF4, ha2b + (size_t)e * FF4 * CDIM, CDIM, h_ab2 + e * CDIM,
         nullptr, pcat_b, 2 * CDIM, MTOK, CDIM, FF4, 0, 1);
    gemm(h2_b, CDIM, hb1b + (size_t)e * CDIM * FF2, FF2, h_bb1 + e * FF2,
         nullptr, tF4_b, FF2, MTOK, FF2, CDIM, 2, 1);
    gemm(tF4_b, FF2, hb2b + (size_t)e * FF2 * CDIM, CDIM, h_bb2 + e * CDIM,
         nullptr, pcat_b + CDIM, 2 * CDIM, MTOK, CDIM, FF2, 0, 1);
    gemm(pcat_b, 2 * CDIM, hpwb + (size_t)e * FF2 * CDIM, CDIM, h_pb + e * CDIM,
         tC_f, nullptr, CDIM, MTOK, CDIM, FF2, 0, 1);
    lnact(tC_f, h2_f, h_ng + e * CDIM, h_nb + e * CDIM, CDIM, 0, y_f, nullptr);
    k_scale_accum<<<gMC, 256, 0, stream>>>(moe, y_f, gates, 6 + e, nMC);
  }

  k_add<<<gMC, 256, 0, stream>>>(x1_f, moe, (float*)d_out, nMC);
}
